// MultiHeadAttention_wpos_78829829751148
// MI455X (gfx1250) — matrix operand order repaired and checked by reading, not compiled, not run on hardware
//
#include <hip/hip_runtime.h>
#include <hip/hip_bf16.h>
#include <math.h>

typedef __attribute__((ext_vector_type(16))) __bf16 v16bf;
typedef __attribute__((ext_vector_type(8)))  __bf16 v8bf;
typedef __attribute__((ext_vector_type(8)))  float  v8f;
typedef unsigned short u16;
typedef unsigned int   u32;

#define L    2048
#define DM   768
#define NH   12
#define DEP  64
#define FEAT 128
#define RLEN 4095   // 2L-1
#define RPAD 4096

// ---------- helpers ----------

__device__ __forceinline__ u16 f2bf(float x) {
  u32 u = __float_as_uint(x);
  u32 r = (u + 0x7FFFu + ((u >> 16) & 1u)) >> 16;   // round-to-nearest-even
  return (u16)r;
}

__device__ __forceinline__ v8bf ld8(const u16* p) {
  return *reinterpret_cast<const v8bf*>(p);
}

__device__ __forceinline__ v16bf cat16(v8bf lo, v8bf hi) {
  v16bf r;
#pragma unroll
  for (int i = 0; i < 8; ++i) { r[i] = lo[i]; r[i + 8] = hi[i]; }
  return r;
}

// 16x16 C tile of A[M,K] * B[K,N], with B supplied transposed: BT[n][k].
// A layout (16-bit A 16x32, ISA 7.12.2): lanes 0-15 hold K {kb..kb+7, kb+16..kb+23},
// lanes 16-31 hold K {kb+8..kb+15, kb+24..kb+31}. B: same K layout as A.
__device__ __forceinline__ v8f wmma_tile_bf(const u16* A, int lda,
                                            const u16* BT, int ldb,
                                            int K, int m0, int n0) {
  const int lane = threadIdx.x & 31;
  const int half = lane >> 4;
  const int lm   = lane & 15;
  const u16* arow = A  + (size_t)(m0 + lm) * lda;
  const u16* brow = BT + (size_t)(n0 + lm) * ldb;
  v8f acc = {};
  for (int kb = 0; kb < K; kb += 32) {
    v8bf alo = ld8(arow + kb + half * 8);
    v8bf ahi = ld8(arow + kb + 16 + half * 8);
    v8bf blo = ld8(brow + kb + half * 8);
    v8bf bhi = ld8(brow + kb + 16 + half * 8);
    acc = __builtin_amdgcn_wmma_f32_16x16x32_bf16(false, cat16(alo, ahi),
                                                  false, cat16(blo, bhi),
                                                  (short)0, acc, false, false);
  }
  return acc;
}

// Same, but A is fp32 in memory, converted to bf16 in flight.
__device__ __forceinline__ v8f wmma_tile_f32a(const float* A, int lda,
                                              const u16* BT, int ldb,
                                              int K, int m0, int n0) {
  const int lane = threadIdx.x & 31;
  const int half = lane >> 4;
  const int lm   = lane & 15;
  const float* arow = A + (size_t)(m0 + lm) * lda;
  const u16* brow = BT + (size_t)(n0 + lm) * ldb;
  v8f acc = {};
  for (int kb = 0; kb < K; kb += 32) {
    v8bf alo, ahi;
#pragma unroll
    for (int i = 0; i < 8; ++i) {
      alo[i] = __builtin_bit_cast(__bf16, f2bf(arow[kb + half * 8 + i]));
      ahi[i] = __builtin_bit_cast(__bf16, f2bf(arow[kb + 16 + half * 8 + i]));
    }
    v8bf blo = ld8(brow + kb + half * 8);
    v8bf bhi = ld8(brow + kb + 16 + half * 8);
    acc = __builtin_amdgcn_wmma_f32_16x16x32_bf16(false, cat16(alo, ahi),
                                                  false, cat16(blo, bhi),
                                                  (short)0, acc, false, false);
  }
  return acc;
}

// ---------- positional encodings ----------

__global__ void k_init_max(u32* maxbits) { if (threadIdx.x == 0) *maxbits = 0u; }

__global__ void k_pe1(u16* pe, float* gprob, u32* maxbits) {
  const int idx = blockIdx.x;      // 0..4095 (4095 = zero pad row)
  const int f   = threadIdx.x;     // 0..127
  __shared__ float sm[FEAT];
  u16* row = pe + (size_t)idx * DM;
  float pr = 0.0f;
  if (idx >= RLEN) {
    row[f] = 0; row[FEAT + f] = 0; row[2 * FEAT + f] = 0;
    row[3 * FEAT + f] = 0; row[4 * FEAT + f] = 0; row[5 * FEAT + f] = 0;
    gprob[(size_t)idx * FEAT + f] = 0.0f;
  } else {
    float p  = (float)(idx - (L - 1));
    float ap = fabsf(p);
    float sg = (p > 0.0f) ? 1.0f : ((p < 0.0f) ? -1.0f : 0.0f);
    // f_exponential: half_life = 2^linspace(3, log2(L)=11, 128)
    float hl = exp2f(3.0f + 8.0f * (float)f / 127.0f);
    float e1 = exp2f(-ap / hl);
    // f_central_mask: cw = 2^(f+1) - 1
    float cw = exp2f((float)(f + 1)) - 1.0f;
    float cm = (cw > ap) ? 1.0f : 0.0f;
    // f_gamma: stdv=8, mean=linspace(16,2048,128)
    float mean = 16.0f + 2032.0f * (float)f / 127.0f;
    float conc = (mean * 0.125f) * (mean * 0.125f);
    float rate = mean * 0.015625f;
    float lp = (conc - 1.0f) * logf(fmaxf(ap, 1e-30f)) - rate * ap
             - (lgammaf(conc) - conc * logf(rate));
    pr = (ap > 0.0f) ? expf(lp) : 0.0f;
    row[f]            = f2bf(e1);
    row[FEAT + f]     = f2bf(e1 * sg);
    row[2 * FEAT + f] = f2bf(cm);
    row[3 * FEAT + f] = f2bf(cm * sg);
    gprob[(size_t)idx * FEAT + f] = pr;
  }
  sm[f] = pr;
  __syncthreads();
  for (int s = FEAT / 2; s > 0; s >>= 1) {
    if (f < s) sm[f] = fmaxf(sm[f], sm[f + s]);
    __syncthreads();
  }
  if (f == 0) atomicMax(maxbits, __float_as_uint(sm[0]));  // probs >= 0
}

__global__ void k_pe2(u16* pe, const float* gprob, const u32* maxbits) {
  const int idx = blockIdx.x;      // 0..4094
  const int f   = threadIdx.x;
  float mx = __uint_as_float(*maxbits);
  float p  = (float)(idx - (L - 1));
  float sg = (p > 0.0f) ? 1.0f : ((p < 0.0f) ? -1.0f : 0.0f);
  float g  = gprob[(size_t)idx * FEAT + f] / mx;
  u16* row = pe + (size_t)idx * DM;
  row[4 * FEAT + f] = f2bf(g);
  row[5 * FEAT + f] = f2bf(g * sg);
}

// ---------- conversions ----------

__global__ void k_wconv(const float* W, u16* WT) {   // WT[n][k] = W[k][n], bf16
  int t = blockIdx.x * blockDim.x + threadIdx.x;
  if (t >= DM * DM) return;
  int n = t / DM, k = t - n * DM;
  WT[t] = f2bf(W[k * DM + n]);
}

__global__ void k_aconv(const float* X, u16* Y, int n) {
  int t = blockIdx.x * blockDim.x + threadIdx.x;
  if (t < n) Y[t] = f2bf(X[t]);
}

// ---------- projections (fused epilogues) ----------

__global__ void k_qproj(const u16* qbf, const u16* WqT, const float* rw,
                        const float* rr, u16* qw, u16* qr) {
  int n0 = blockIdx.x * 16, m0 = blockIdx.y * 16;
  v8f acc = wmma_tile_bf(qbf, DM, WqT, DM, DM, m0, n0);
  int lane = threadIdx.x & 31, half = lane >> 4, lm = lane & 15;
  int nabs = n0 + lm, h = nabs >> 6, d = nabs & 63;
  float brw = rw[nabs], brr = rr[nabs];
#pragma unroll
  for (int j = 0; j < 8; ++j) {
    int l = m0 + j + 8 * half;
    float v = acc[j] * 0.125f;                 // / sqrt(DEPTH)
    size_t o = ((size_t)(h * L + l)) * DEP + d;
    qw[o] = f2bf(v + brw);
    qr[o] = f2bf(v + brr);
  }
}

__global__ void k_kproj(const u16* kbf, const u16* WkT, u16* kh) {
  int n0 = blockIdx.x * 16, m0 = blockIdx.y * 16;
  v8f acc = wmma_tile_bf(kbf, DM, WkT, DM, DM, m0, n0);
  int lane = threadIdx.x & 31, half = lane >> 4, lm = lane & 15;
  int nabs = n0 + lm, h = nabs >> 6, d = nabs & 63;
#pragma unroll
  for (int j = 0; j < 8; ++j) {
    int l = m0 + j + 8 * half;
    kh[((size_t)(h * L + l)) * DEP + d] = f2bf(acc[j]);
  }
}

__global__ void k_vproj(const u16* vbf, const u16* WvT, u16* vhT) {
  int n0 = blockIdx.x * 16, m0 = blockIdx.y * 16;
  v8f acc = wmma_tile_bf(vbf, DM, WvT, DM, DM, m0, n0);
  int lane = threadIdx.x & 31, half = lane >> 4, lm = lane & 15;
  int nabs = n0 + lm, h = nabs >> 6, d = nabs & 63;
#pragma unroll
  for (int j = 0; j < 8; ++j) {
    int l = m0 + j + 8 * half;
    vhT[((size_t)(h * DEP + d)) * L + l] = f2bf(acc[j]);   // [H][D][L]
  }
}

__global__ void k_rkproj(const u16* pe, const u16* WrkT, u16* rk) {
  int n0 = blockIdx.x * 16, m0 = blockIdx.y * 16;          // m tiles: 256 (4096 rows)
  v8f acc = wmma_tile_bf(pe, DM, WrkT, DM, DM, m0, n0);
  int lane = threadIdx.x & 31, half = lane >> 4, lm = lane & 15;
  int nabs = n0 + lm, h = nabs >> 6, d = nabs & 63;
#pragma unroll
  for (int j = 0; j < 8; ++j) {
    int r = m0 + j + 8 * half;
    rk[((size_t)h * RPAD + r) * DEP + d] = f2bf(acc[j]);   // [H][4096][D]
  }
}

// ---------- logits ----------

__global__ void k_content(const u16* qw, const u16* kh, float* attn) {
  int h = blockIdx.z, q0 = blockIdx.y * 16, k0 = blockIdx.x * 16;
  v8f acc = wmma_tile_bf(qw + (size_t)h * L * DEP, DEP,
                         kh + (size_t)h * L * DEP, DEP, DEP, q0, k0);
  int lane = threadIdx.x & 31, half = lane >> 4, lm = lane & 15;
  float* out = attn + (size_t)h * L * L;
#pragma unroll
  for (int j = 0; j < 8; ++j) {
    int q = q0 + j + 8 * half;
    out[(size_t)q * L + k0 + lm] = acc[j];
  }
}

// relative logits with relative_shift fused: shifted[q,k] = rel[q, k - q + (L-1)]
__global__ void k_rel(const u16* qr, const u16* rk, float* attn) {
  int h = blockIdx.z, q0 = blockIdx.y * 16, r0 = blockIdx.x * 16;
  int kmin = r0 + q0 - (L - 1);
  if (kmin + 30 < 0 || kmin > L - 1) return;   // tile maps entirely out of [0,L)
  v8f acc = wmma_tile_bf(qr + (size_t)h * L * DEP, DEP,
                         rk + (size_t)h * RPAD * DEP, DEP, DEP, q0, r0);
  int lane = threadIdx.x & 31, half = lane >> 4, lm = lane & 15;
  float* out = attn + (size_t)h * L * L;
  int r = r0 + lm;
#pragma unroll
  for (int j = 0; j < 8; ++j) {
    int q = q0 + j + 8 * half;
    int k = r + q - (L - 1);
    if (k >= 0 && k < L) out[(size_t)q * L + k] += acc[j];
  }
}

// ---------- softmax (mask==1 -> -10000), in place ----------

__global__ void k_softmax(float* attn, const int* mask) {
  int h = blockIdx.y, q = blockIdx.x;
  float* row = attn + ((size_t)h * L + q) * L;
  __shared__ float sm[256];
  int t = threadIdx.x;
  float mx = -3.4e38f;
  for (int k = t; k < L; k += 256) {
    float v = (mask[k] == 1) ? -10000.0f : row[k];
    mx = fmaxf(mx, v);
  }
  sm[t] = mx; __syncthreads();
  for (int s = 128; s > 0; s >>= 1) { if (t < s) sm[t] = fmaxf(sm[t], sm[t + s]); __syncthreads(); }
  mx = sm[0]; __syncthreads();
  float sum = 0.0f;
  for (int k = t; k < L; k += 256) {
    float v = (mask[k] == 1) ? -10000.0f : row[k];
    float e = expf(v - mx);
    row[k] = e;
    sum += e;
  }
  sm[t] = sum; __syncthreads();
  for (int s = 128; s > 0; s >>= 1) { if (t < s) sm[t] += sm[t + s]; __syncthreads(); }
  float inv = 1.0f / sm[0];
  for (int k = t; k < L; k += 256) row[k] *= inv;
}

// ---------- attn x V and output projection ----------

__global__ void k_attended(const float* attn, const u16* vhT, u16* attC) {
  int h = blockIdx.z, q0 = blockIdx.y * 16, d0 = blockIdx.x * 16;
  v8f acc = wmma_tile_f32a(attn + (size_t)h * L * L, L,
                           vhT + (size_t)h * DEP * L, L, L, q0, d0);
  int lane = threadIdx.x & 31, half = lane >> 4, lm = lane & 15;
#pragma unroll
  for (int j = 0; j < 8; ++j) {
    int q = q0 + j + 8 * half;
    attC[(size_t)q * DM + h * DEP + d0 + lm] = f2bf(acc[j]);  // concat layout
  }
}

__global__ void k_outproj(const u16* attC, const u16* WoT, const float* bo, float* out) {
  int n0 = blockIdx.x * 16, m0 = blockIdx.y * 16;
  v8f acc = wmma_tile_bf(attC, DM, WoT, DM, DM, m0, n0);
  int lane = threadIdx.x & 31, half = lane >> 4, lm = lane & 15;
  float b = bo[n0 + lm];
#pragma unroll
  for (int j = 0; j < 8; ++j) {
    int m = m0 + j + 8 * half;
    out[(size_t)m * DM + n0 + lm] = acc[j] + b;
  }
}

// ---------- launch ----------

extern "C" void kernel_launch(void* const* d_in, const int* in_sizes, int n_in,
                              void* d_out, int out_size, void* d_ws, size_t ws_size,
                              hipStream_t stream) {
  (void)in_sizes; (void)n_in; (void)out_size; (void)ws_size;
  const float* v_in = (const float*)d_in[0];
  const float* k_in = (const float*)d_in[1];
  const float* q_in = (const float*)d_in[2];
  const int*   mask = (const int*)d_in[3];
  const float* Wq   = (const float*)d_in[4];
  const float* Wk   = (const float*)d_in[5];
  const float* Wv   = (const float*)d_in[6];
  const float* Wrk  = (const float*)d_in[7];
  const float* rw   = (const float*)d_in[8];
  const float* rr   = (const float*)d_in[9];
  const float* Wo   = (const float*)d_in[10];
  const float* bo   = (const float*)d_in[11];

  char* ws = (char*)d_ws;
  size_t off = 0;
  auto carve = [&](size_t bytes) -> void* {
    void* p = ws + off;
    off = (off + bytes + 255) & ~(size_t)255;
    return p;
  };
  u16*   pe    = (u16*)  carve((size_t)RPAD * DM * 2);
  float* gprob = (float*)carve((size_t)RPAD * FEAT * 4);
  u32*   mxb   = (u32*)  carve(256);
  u16*   qbf   = (u16*)  carve((size_t)L * DM * 2);
  u16*   kbf   = (u16*)  carve((size_t)L * DM * 2);
  u16*   vbf   = (u16*)  carve((size_t)L * DM * 2);
  u16*   WqT   = (u16*)  carve((size_t)DM * DM * 2);
  u16*   WkT   = (u16*)  carve((size_t)DM * DM * 2);
  u16*   WvT   = (u16*)  carve((size_t)DM * DM * 2);
  u16*   WrkT  = (u16*)  carve((size_t)DM * DM * 2);
  u16*   WoT   = (u16*)  carve((size_t)DM * DM * 2);
  u16*   qw    = (u16*)  carve((size_t)NH * L * DEP * 2);
  u16*   qr    = (u16*)  carve((size_t)NH * L * DEP * 2);
  u16*   kh    = (u16*)  carve((size_t)NH * L * DEP * 2);
  u16*   vhT   = (u16*)  carve((size_t)NH * DEP * L * 2);
  u16*   rk    = (u16*)  carve((size_t)NH * RPAD * DEP * 2);
  u16*   attC  = (u16*)  carve((size_t)L * DM * 2);

  float* outp = (float*)d_out;                 // [L, DM]
  float* attn = outp + (size_t)L * DM;         // [NH, L, L] attention map (also logits scratch)

  // positional encodings
  k_init_max<<<1, 32, 0, stream>>>(mxb);
  k_pe1<<<RPAD, FEAT, 0, stream>>>(pe, gprob, mxb);
  k_pe2<<<RLEN, FEAT, 0, stream>>>(pe, gprob, mxb);

  // conversions
  int wblocks = (DM * DM + 255) / 256;
  k_wconv<<<wblocks, 256, 0, stream>>>(Wq,  WqT);
  k_wconv<<<wblocks, 256, 0, stream>>>(Wk,  WkT);
  k_wconv<<<wblocks, 256, 0, stream>>>(Wv,  WvT);
  k_wconv<<<wblocks, 256, 0, stream>>>(Wrk, WrkT);
  k_wconv<<<wblocks, 256, 0, stream>>>(Wo,  WoT);
  int ablocks = (L * DM + 255) / 256;
  k_aconv<<<ablocks, 256, 0, stream>>>(q_in, qbf, L * DM);
  k_aconv<<<ablocks, 256, 0, stream>>>(k_in, kbf, L * DM);
  k_aconv<<<ablocks, 256, 0, stream>>>(v_in, vbf, L * DM);

  // projections
  k_qproj <<<dim3(DM / 16, L / 16),    32, 0, stream>>>(qbf, WqT, rw, rr, qw, qr);
  k_kproj <<<dim3(DM / 16, L / 16),    32, 0, stream>>>(kbf, WkT, kh);
  k_vproj <<<dim3(DM / 16, L / 16),    32, 0, stream>>>(vbf, WvT, vhT);
  k_rkproj<<<dim3(DM / 16, RPAD / 16), 32, 0, stream>>>(pe, WrkT, rk);

  // logits, softmax
  k_content<<<dim3(L / 16, L / 16, NH),    32, 0, stream>>>(qw, kh, attn);
  k_rel    <<<dim3(RPAD / 16, L / 16, NH), 32, 0, stream>>>(qr, rk, attn);
  k_softmax<<<dim3(L, NH), 256, 0, stream>>>(attn, mask);

  // attn x V, output projection
  k_attended<<<dim3(DEP / 16, L / 16, NH), 32, 0, stream>>>(attn, vhT, attC);
  k_outproj <<<dim3(DM / 16, L / 16),      32, 0, stream>>>(attC, WoT, bo, outp);
}
